// GCU_37306085933363
// MI455X (gfx1250) — matrix operand order repaired and checked by reading, not compiled, not run on hardware
//
#include <hip/hip_runtime.h>

typedef __attribute__((ext_vector_type(16))) _Float16 v16h;
typedef __attribute__((ext_vector_type(8)))  _Float16 v8h;
typedef __attribute__((ext_vector_type(4)))  _Float16 v4h;
typedef __attribute__((ext_vector_type(8)))  float    v8f;

#define N_NODES 100000
#define N_EDGES 1600000
#define E_TOT   (N_EDGES + N_NODES)     /* 1,700,000 */
#define TILES_PER_CONV (E_TOT / 16)     /* 106,250 exact */
#define NODE_TILES (N_NODES / 16)       /* 6,250 exact */
#define BN_EPS 1e-5f

// ---------- helpers ----------
__device__ __forceinline__ v8f wmma16(v16h a, v16h b, v8f c) {
    return __builtin_amdgcn_wmma_f32_16x16x32_f16(false, a, false, b, (short)0, c,
                                                  false, false);
}

__device__ __forceinline__ v16h load16p(const _Float16* p0, const _Float16* p1) {
    union { v8h h[2]; v16h v; } u;
    u.h[0] = *(const v8h*)p0;
    u.h[1] = *(const v8h*)p1;
    return u.v;
}
__device__ __forceinline__ v16h load16(const _Float16* p) { return load16p(p, p + 8); }

// monotone-int float atomic max (order independent -> deterministic)
__device__ __forceinline__ void atomicMaxF(float* a, float v) {
    if (v >= 0.f) atomicMax((int*)a, __float_as_int(v));
    else          atomicMin((unsigned int*)a, __float_as_uint(v));
}

// ---------- workspace layout ----------
// floats [0..63] scale1[k][32] | [64..127] shift1 | [128..191] scale2 | [192..255] shift2
//        [256..319] fscale[64] | [320..383] fshift[64]
// halves at byte 1536: W1frag 4096 | W2frag 2048 | mWfrag 4096
//   W1frag: i = k*2048 + kt*1024 + nt*512 + lane*16 + j
//   W2frag: i = k*1024 + nt*512 + lane*16 + j
//   mWfrag: i = kt*2048 + nt*512 + lane*16 + j
//   fragment element = W[K][N] with K = kt*32 + (lane/16)*16 + j, N = nt*16 + lane%16

__global__ void prep_kernel(const float* __restrict__ ecW1,
                            const float* __restrict__ ecg1, const float* __restrict__ ecbt1,
                            const float* __restrict__ ecm1, const float* __restrict__ ecv1,
                            const float* __restrict__ ecW2,
                            const float* __restrict__ ecg2, const float* __restrict__ ecbt2,
                            const float* __restrict__ ecm2, const float* __restrict__ ecv2,
                            const float* __restrict__ mW,
                            const float* __restrict__ mg, const float* __restrict__ mbt,
                            const float* __restrict__ mm, const float* __restrict__ mv,
                            float* __restrict__ aff, _Float16* __restrict__ frags) {
    int t = threadIdx.x;
    for (int i = t; i < 64; i += blockDim.x) {
        float s1 = ecg1[i] * rsqrtf(ecv1[i] + BN_EPS);
        aff[i] = s1; aff[64 + i] = ecbt1[i] - ecm1[i] * s1;
        float s2 = ecg2[i] * rsqrtf(ecv2[i] + BN_EPS);
        aff[128 + i] = s2; aff[192 + i] = ecbt2[i] - ecm2[i] * s2;
        float sf = mg[i] * rsqrtf(mv[i] + BN_EPS);
        aff[256 + i] = sf; aff[320 + i] = mbt[i] - mm[i] * sf;
    }
    _Float16* w1f = frags;
    _Float16* w2f = frags + 4096;
    _Float16* mwf = frags + 6144;
    for (int i = t; i < 4096; i += blockDim.x) {          // W1 [2][64][32]
        int j = i & 15, lane = (i >> 4) & 31, nt = (i >> 9) & 1;
        int kt = (i >> 10) & 1, k = (i >> 11) & 1;
        int K = kt * 32 + (lane >> 4) * 16 + j;
        int Nn = nt * 16 + (lane & 15);
        w1f[i] = (_Float16)ecW1[(k * 64 + K) * 32 + Nn];
    }
    for (int i = t; i < 2048; i += blockDim.x) {          // W2 [2][32][32]
        int j = i & 15, lane = (i >> 4) & 31, nt = (i >> 9) & 1, k = (i >> 10) & 1;
        int K = (lane >> 4) * 16 + j;
        int Nn = nt * 16 + (lane & 15);
        w2f[i] = (_Float16)ecW2[(k * 32 + K) * 32 + Nn];
    }
    for (int i = t; i < 4096; i += blockDim.x) {          // mW [64][64]
        int j = i & 15, lane = (i >> 4) & 31, nt = (i >> 9) & 3, kt = (i >> 11) & 1;
        int K = kt * 32 + (lane >> 4) * 16 + j;
        int Nn = nt * 16 + (lane & 15);
        mwf[i] = (_Float16)mW[K * 64 + Nn];
    }
}

__global__ void init_kernel(float* __restrict__ out, int n) {
    for (int i = blockIdx.x * blockDim.x + threadIdx.x; i < n; i += gridDim.x * blockDim.x)
        out[i] = -3.4028234663852886e+38f;
}

// one wave == one 16-edge WMMA tile
__global__ void __launch_bounds__(128) edge_kernel(
        const float* __restrict__ x,
        const int* __restrict__ ei1, const int* __restrict__ ei2,
        const float* __restrict__ ecb1, const float* __restrict__ ecb2,
        const float* __restrict__ aff, const _Float16* __restrict__ frags,
        float* __restrict__ out) {
    __shared__ _Float16 s_in[4][16 * 64];   // [edge][ xi(32) | xj-xi(32) ] f16
    __shared__ _Float16 s_h[4][16 * 32];    // hidden activations
    __shared__ int      s_dst[4][16];

    const int w    = threadIdx.x >> 5;
    const int lane = threadIdx.x & 31;
    const int gw   = blockIdx.x * 4 + w;
    const int k    = gw / TILES_PER_CONV;              // conv 0/1 (wave-uniform)
    const int tile = gw - k * TILES_PER_CONV;
    const int base = tile * 16;
    const int* ei  = k ? ei2 : ei1;

    // ---- gather 16 edges: 2 lanes per edge, 16 channels each ----
    {
        const int m = lane >> 1, half = lane & 1;
        const int e = base + m;
        int s, d;
        if (e < N_EDGES) { s = ei[e]; d = ei[N_EDGES + e]; }
        else             { s = e - N_EDGES; d = s; }     // self loop
        if (half == 0) s_dst[w][m] = d;
        const float4* xi = (const float4*)(x + d * 32 + half * 16);
        const float4* xj = (const float4*)(x + s * 32 + half * 16);
#pragma unroll
        for (int c4 = 0; c4 < 4; ++c4) {
            float4 a = xi[c4];
            float4 b = xj[c4];
            v4h ha = {(_Float16)a.x, (_Float16)a.y, (_Float16)a.z, (_Float16)a.w};
            v4h hd = {(_Float16)(b.x - a.x), (_Float16)(b.y - a.y),
                      (_Float16)(b.z - a.z), (_Float16)(b.w - a.w)};
            *(v4h*)&s_in[w][m * 64 + half * 16 + c4 * 4]      = ha;
            *(v4h*)&s_in[w][m * 64 + 32 + half * 16 + c4 * 4] = hd;
        }
    }

    // ---- per-lane fused BN/bias params (channel = nt*16 + lane%16) ----
    const int nlo = lane & 15;
    const float b1v0 = ecb1[k * 32 + nlo],        b1v1 = ecb1[k * 32 + 16 + nlo];
    const float sc10 = aff[k * 32 + nlo],         sc11 = aff[k * 32 + 16 + nlo];
    const float sh10 = aff[64 + k * 32 + nlo],    sh11 = aff[64 + k * 32 + 16 + nlo];
    const float b2v0 = ecb2[k * 32 + nlo],        b2v1 = ecb2[k * 32 + 16 + nlo];
    const float sc20 = aff[128 + k * 32 + nlo],   sc21 = aff[128 + k * 32 + 16 + nlo];
    const float sh20 = aff[192 + k * 32 + nlo],   sh21 = aff[192 + k * 32 + 16 + nlo];

    // ---- weight B-fragments (pre-swizzled) ----
    const _Float16* w1f = frags;
    const _Float16* w2f = frags + 4096;
    const v16h b1_00 = load16(w1f + k * 2048 + 0 * 1024 + 0 * 512 + lane * 16);
    const v16h b1_01 = load16(w1f + k * 2048 + 0 * 1024 + 1 * 512 + lane * 16);
    const v16h b1_10 = load16(w1f + k * 2048 + 1 * 1024 + 0 * 512 + lane * 16);
    const v16h b1_11 = load16(w1f + k * 2048 + 1 * 1024 + 1 * 512 + lane * 16);
    const v16h b2_0  = load16(w2f + k * 1024 + 0 * 512 + lane * 16);
    const v16h b2_1  = load16(w2f + k * 1024 + 1 * 512 + lane * 16);

    // ---- A fragments, contiguous k = 16*(lane/16) + s, same order as B frags ----
    const int row = lane & 15;
    const int cb  = (lane >> 4) * 16;                // K-half selector
    const _Float16* rp = &s_in[w][row * 64];
    const v16h a_lo = load16p(rp + cb,      rp + cb + 8);         // K 0..31
    const v16h a_hi = load16p(rp + 32 + cb, rp + 32 + cb + 8);    // K 32..63

    // ---- GEMM1: [16x64] @ [64x32] ----
    v8f c0 = {}, c1 = {};
    c0 = wmma16(a_lo, b1_00, c0);
    c0 = wmma16(a_hi, b1_10, c0);
    c1 = wmma16(a_lo, b1_01, c1);
    c1 = wmma16(a_hi, b1_11, c1);

    // ---- epilogue 1: bias+ReLU+BN -> f16 LDS (row M = r + 8*(lane/16)) ----
    const int rbase = (lane >> 4) * 8;
#pragma unroll
    for (int r = 0; r < 8; ++r) {
        int M = r + rbase;
        float h0 = fmaxf(c0[r] + b1v0, 0.f) * sc10 + sh10;
        float h1 = fmaxf(c1[r] + b1v1, 0.f) * sc11 + sh11;
        s_h[w][M * 32 + nlo]      = (_Float16)h0;
        s_h[w][M * 32 + 16 + nlo] = (_Float16)h1;
    }

    // ---- GEMM2: [16x32] @ [32x32] ----
    const _Float16* hp = &s_h[w][row * 32];
    const v16h a2 = load16p(hp + cb, hp + cb + 8);
    v8f d0 = {}, d1 = {};
    d0 = wmma16(a2, b2_0, d0);
    d1 = wmma16(a2, b2_1, d1);

    // ---- epilogue 2: bias+ReLU+BN, scatter-max into out[:, k*32 + n] ----
#pragma unroll
    for (int r = 0; r < 8; ++r) {
        int M  = r + rbase;
        int dn = s_dst[w][M];
        float v0 = fmaxf(d0[r] + b2v0, 0.f) * sc20 + sh20;
        float v1 = fmaxf(d1[r] + b2v1, 0.f) * sc21 + sh21;
        atomicMaxF(out + dn * 64 + k * 32 + nlo,      v0);
        atomicMaxF(out + dn * 64 + k * 32 + 16 + nlo, v1);
    }
}

// final node-wise 64->64 Linear + ReLU + BN, in place on out (one wave == 16 nodes)
__global__ void __launch_bounds__(128) node_kernel(
        const float* __restrict__ mb, const float* __restrict__ aff,
        const _Float16* __restrict__ mwf, float* __restrict__ out) {
    __shared__ _Float16 s_a[4][16 * 64];
    const int w = threadIdx.x >> 5, lane = threadIdx.x & 31;
    const int gw = blockIdx.x * 4 + w;
    if (gw >= NODE_TILES) return;                     // wave-uniform
    const int base = gw * 16;

    {   // gather 16 rows of out -> f16 LDS
        const int m = lane >> 1, half = lane & 1;
        const float4* rp = (const float4*)(out + (base + m) * 64 + half * 32);
#pragma unroll
        for (int c4 = 0; c4 < 8; ++c4) {
            float4 a = rp[c4];
            v4h hv = {(_Float16)a.x, (_Float16)a.y, (_Float16)a.z, (_Float16)a.w};
            *(v4h*)&s_a[w][m * 64 + half * 32 + c4 * 4] = hv;
        }
    }

    const int nlo = lane & 15, row = lane & 15;
    const int cb = (lane >> 4) * 16, rbase = (lane >> 4) * 8;
    const _Float16* rp = &s_a[w][row * 64];
    const v16h a0 = load16p(rp + cb,      rp + cb + 8);        // K 0..31
    const v16h a1 = load16p(rp + 32 + cb, rp + 32 + cb + 8);   // K 32..63

    v8f acc[4];
#pragma unroll
    for (int nt = 0; nt < 4; ++nt) {
        v16h f0 = load16(mwf + 0 * 2048 + nt * 512 + lane * 16);
        v16h f1 = load16(mwf + 1 * 2048 + nt * 512 + lane * 16);
        v8f c = {};
        c = wmma16(a0, f0, c);
        c = wmma16(a1, f1, c);
        acc[nt] = c;
    }
#pragma unroll
    for (int nt = 0; nt < 4; ++nt) {
        int n = nt * 16 + nlo;
        float bb = mb[n], ss = aff[256 + n], tt = aff[320 + n];
#pragma unroll
        for (int r = 0; r < 8; ++r) {
            int node = base + r + rbase;
            out[node * 64 + n] = fmaxf(acc[nt][r] + bb, 0.f) * ss + tt;
        }
    }
}

extern "C" void kernel_launch(void* const* d_in, const int* in_sizes, int n_in,
                              void* d_out, int out_size, void* d_ws, size_t ws_size,
                              hipStream_t stream) {
    const float* x    = (const float*)d_in[0];
    const int*   ei1  = (const int*)d_in[1];
    const int*   ei2  = (const int*)d_in[2];
    const float* ecW1 = (const float*)d_in[3];
    const float* ecb1 = (const float*)d_in[4];
    const float* ecg1 = (const float*)d_in[5];
    const float* ecbt1= (const float*)d_in[6];
    const float* ecm1 = (const float*)d_in[7];
    const float* ecv1 = (const float*)d_in[8];
    const float* ecW2 = (const float*)d_in[9];
    const float* ecb2 = (const float*)d_in[10];
    const float* ecg2 = (const float*)d_in[11];
    const float* ecbt2= (const float*)d_in[12];
    const float* ecm2 = (const float*)d_in[13];
    const float* ecv2 = (const float*)d_in[14];
    const float* mW   = (const float*)d_in[15];
    const float* mb   = (const float*)d_in[16];
    const float* mg   = (const float*)d_in[17];
    const float* mbt  = (const float*)d_in[18];
    const float* mm   = (const float*)d_in[19];
    const float* mv   = (const float*)d_in[20];

    float*     out   = (float*)d_out;                        // [N, 64]
    float*     aff   = (float*)d_ws;                         // 384 floats
    _Float16*  frags = (_Float16*)((char*)d_ws + 1536);      // 10240 halves

    prep_kernel<<<1, 256, 0, stream>>>(ecW1, ecg1, ecbt1, ecm1, ecv1,
                                       ecW2, ecg2, ecbt2, ecm2, ecv2,
                                       mW, mg, mbt, mm, mv, aff, frags);
    init_kernel<<<4096, 256, 0, stream>>>(out, N_NODES * 64);

    // 2 convs * 106,250 tiles, 4 waves (tiles) per 128-thread block -> 53,125 blocks
    edge_kernel<<<(2 * TILES_PER_CONV) / 4, 128, 0, stream>>>(
        x, ei1, ei2, ecb1, ecb2, aff, frags, out);

    node_kernel<<<(NODE_TILES + 3) / 4, 128, 0, stream>>>(mb, aff, frags + 6144, out);
}
